// lstm_8263517078097
// MI455X (gfx1250) — matrix operand order repaired and checked by reading, not compiled, not run on hardware
//
#include <hip/hip_runtime.h>

typedef __attribute__((ext_vector_type(16))) __bf16 v16bf;
typedef __attribute__((ext_vector_type(8)))  __bf16 v8bf;
typedef __attribute__((ext_vector_type(4)))  __bf16 v4bf;
typedef __attribute__((ext_vector_type(8)))  float  v8f;
typedef __attribute__((ext_vector_type(4)))  float  v4f;

#define B_    8192
#define T_    204
#define I_    5
#define H_    24
#define L_    10
#define O_    612
#define G4    96      // 4*H gate rows
#define K_    32      // padded K: [h(24) | x(5) | bias(1) | pad(2)]
#define KX_   24      // x slot base in K
#define KB_   29      // bias row index in K
#define WAVES 4
#define TILE  16      // batch columns per wave
#define TB    4       // timesteps per fc0 register-staging block (204 = 51*4)

// branch-free activations: v_exp_f32 + v_rcp_f32, correct saturation at +-inf
__device__ __forceinline__ float fast_sigmoid(float x) {
  const float e = __builtin_amdgcn_exp2f(x * -1.4426950408889634f);
  return __builtin_amdgcn_rcpf(1.f + e);
}
__device__ __forceinline__ float fast_tanh(float x) {
  const float e = __builtin_amdgcn_exp2f(x * -2.8853900817779268f);
  return 2.f * __builtin_amdgcn_rcpf(1.f + e) - 1.f;
}

__global__ __launch_bounds__(128) void lstm_fused_kernel(
    const float* __restrict__ x,
    const float* __restrict__ W_ih,
    const float* __restrict__ W_hh,
    const float* __restrict__ b_ih,
    const float* __restrict__ b_hh,
    const float* __restrict__ fc0_w,
    const float* __restrict__ fc0_b,
    const float* __restrict__ out_w,
    const float* __restrict__ out_b,
    float* __restrict__ out)
{
  __shared__ __align__(16) __bf16 sA[WAVES][TILE * K_];   // hcat staging [b][k]
  __shared__ __align__(16) float  sG[WAVES][TILE * G4];   // gates^T: [b][96]
  __shared__ float  sAct[WAVES][TILE * L_];               // relu activations
  __shared__ float  sOW[O_ * L_];
  __shared__ float  sOB[O_];

  const int tid    = threadIdx.x;
  const int wave   = tid >> 5;
  const int lane   = tid & 31;
  const int lane16 = lane & 15;
  const int hi     = lane >> 4;
  const int tileBase = (blockIdx.x * WAVES + wave) * TILE;

  // ---- block-wide one-time preload of output layer into LDS ----
  for (int e = tid; e < O_ * L_; e += 128) sOW[e] = out_w[e];
  for (int e = tid; e < O_;      e += 128) sOB[e] = out_b[e];
  __syncthreads();

  // ---- loop-invariant gate A-tiles (weights as A-matrix, bf16) ----
  // row n = 16*t6 + lane16 (gate index); K: k<24 -> W_hh[n][k],
  // k in [24,29) -> W_ih[n][k-24], k==29 -> b_ih[n]+b_hh[n], else 0
  const int aK1 = hi ? 8 : 0;   // A-layout K-chunk base per lane half
  v16bf wA6[6];
  #pragma unroll
  for (int t6 = 0; t6 < 6; ++t6) {
    const int n = 16 * t6 + lane16;
    #pragma unroll
    for (int e = 0; e < 16; ++e) {
      const int k = (e < 8) ? (aK1 + e) : (aK1 + 16 + (e - 8));
      float w = 0.f;
      if (k < H_)            w = W_hh[n * H_ + k];
      else if (k < H_ + I_)  w = W_ih[n * I_ + (k - H_)];
      else if (k == KB_)     w = b_ih[n] + b_hh[n];
      wA6[t6][e] = (__bf16)w;
    }
  }

  __bf16* wAs = sA[wave];
  float*  wG  = sG[wave];

  // zero hcat staging (h0 = 0, pad = 0); plant bias-row 1.0 (dup writes benign)
  for (int e = lane; e < TILE * K_; e += 32) wAs[e] = (__bf16)0.f;
  wAs[lane16 * K_ + KB_] = (__bf16)1.f;
  asm volatile("" ::: "memory");

  // per-lane cell state: batch m = lane16, hidden n = hi*12 + u
  float cst[12];
  #pragma unroll
  for (int u = 0; u < 12; ++u) cst[u] = 0.f;

  // fc0 accumulator lives in WMMA C/D registers across all timesteps
  v8f accFC = {};

  const int  m_my    = lane16;
  const bool fcValid = (lane16 < L_);
  const float* fwRow = fc0_w + (size_t)(fcValid ? lane16 : (L_ - 1)) * (H_ * T_);

  // branch-free x staging plan: 80 elems per step, 3 per lane (clamped dup)
  const float* xPtr[3];
  int xDst[3];
  #pragma unroll
  for (int j = 0; j < 3; ++j) {
    int e = lane + 32 * j; e = e < (TILE * I_) ? e : (TILE * I_ - 1);
    const int m = e / I_, i = e - m * I_;
    xPtr[j] = x + (size_t)(tileBase + m) * (T_ * I_) + i;
    xDst[j] = m * K_ + KX_ + i;
  }

  // software pipeline: x values for step t are loaded during step t-1
  float xv[3];
  #pragma unroll
  for (int j = 0; j < 3; ++j) xv[j] = xPtr[j][0];

  for (int tb = 0; tb < T_ / TB; ++tb) {
    // ---- stage fc0 A-fragments for TB steps into registers (branch-free) ----
    v16bf fcA[TB];
    #pragma unroll
    for (int ts = 0; ts < TB; ++ts) {
      const float* fw = fwRow + (tb * TB + ts) * H_;
      v4f f4[6];
      #pragma unroll
      for (int q = 0; q < 6; ++q) f4[q] = *(const v4f*)(fw + 4 * q);
      #pragma unroll
      for (int e = 0; e < 16; ++e) {
        // hi=0: k=e (e<8) else k=e+8 ; hi=1: k=8+e (e<8) else k>=24 -> 0
        const float wlo = (e < 8) ? f4[e >> 2][e & 3] : f4[(e + 8) >> 2][(e + 8) & 3];
        const float whi = (e < 8) ? f4[(e + 8) >> 2][(e + 8) & 3] : 0.f;
        float w = hi ? whi : wlo;
        w = fcValid ? w : 0.f;
        fcA[ts][e] = (__bf16)w;
      }
    }

    // fully unrolled step quad: static fcA indexing (no v_movrels), wide window
    #pragma unroll
    for (int ts = 0; ts < TB; ++ts) {
      const int t = tb * TB + ts;

      // commit prefetched x (no load wait), then issue next step's loads
      #pragma unroll
      for (int j = 0; j < 3; ++j) wAs[xDst[j]] = (__bf16)xv[j];
      const int tn = (t + 1 < T_) ? (t + 1) : t;   // clamped prefetch
      #pragma unroll
      for (int j = 0; j < 3; ++j) xv[j] = xPtr[j][(size_t)tn * I_];
      asm volatile("" ::: "memory");

      // build B fragment (hcat column = this lane's batch): two 16B LDS loads
      const v8bf b_lo = *(const v8bf*)&wAs[m_my * K_ + hi * 8];
      const v8bf b_hi = *(const v8bf*)&wAs[m_my * K_ + 16 + hi * 8];
      v16bf bM = __builtin_shufflevector(b_lo, b_hi,
          0,1,2,3,4,5,6,7,8,9,10,11,12,13,14,15);

      // gates^T[96x16] = W[96x32] * hcat^T[32x16] (+bias via K-row 29)
      #pragma unroll
      for (int t6 = 0; t6 < 6; ++t6) {
        v8f acc = {};
        acc = __builtin_amdgcn_wmma_f32_16x16x32_bf16(
            false, wA6[t6], false, bM, (short)0, acc, false, false);
        // lane: batch col m=lane16, rows g = 16*t6 + hi*8 + j  (consecutive!)
        float* dst = &wG[m_my * G4 + 16 * t6 + hi * 8];
        *(v4f*)(dst + 0) = __builtin_shufflevector(acc, acc, 0, 1, 2, 3);
        *(v4f*)(dst + 4) = __builtin_shufflevector(acc, acc, 4, 5, 6, 7);
      }

      // fc0 partial: accFC += fc0W * hcat^T (x/bias K-rows hit zero; C=D in place)
      accFC = __builtin_amdgcn_wmma_f32_16x16x32_bf16(
          false, fcA[ts], false, bM, (short)0, accFC, false, false);
      asm volatile("" ::: "memory");

      // pointwise update: batch m=lane16, n = hi*12+u ; all-vector LDS traffic
      {
        const float* g0 = &wG[m_my * G4 + hi * 12];
        v4f vi[3], vf[3], vg[3], vo[3];
        #pragma unroll
        for (int q = 0; q < 3; ++q) {
          vi[q] = *(const v4f*)(g0 +      4 * q);
          vf[q] = *(const v4f*)(g0 + 24 + 4 * q);
          vg[q] = *(const v4f*)(g0 + 48 + 4 * q);
          vo[q] = *(const v4f*)(g0 + 72 + 4 * q);
        }
        v4bf hv[3];
        #pragma unroll
        for (int u = 0; u < 12; ++u) {
          const float gi = vi[u >> 2][u & 3];
          const float gf = vf[u >> 2][u & 3];
          const float gg = vg[u >> 2][u & 3];
          const float go = vo[u >> 2][u & 3];
          const float c  = fast_sigmoid(gf) * cst[u] + fast_sigmoid(gi) * fast_tanh(gg);
          cst[u] = c;
          const float h  = fast_sigmoid(go) * fast_tanh(c);
          hv[u >> 2][u & 3] = (__bf16)h;
        }
        // h writeback: 3 aligned 8B stores into hcat row (k = hi*12 .. +11)
        __bf16* hdst = &wAs[m_my * K_ + hi * 12];
        #pragma unroll
        for (int q = 0; q < 3; ++q) *(v4bf*)(hdst + 4 * q) = hv[q];
      }
      asm volatile("" ::: "memory");
    }
  }

  // ---- fc0 epilogue: lane holds batch col m=lane16, rows l = hi*8+j ----
  float* wAct = sAct[wave];
  #pragma unroll
  for (int j = 0; j < 8; ++j) {
    const int l = hi * 8 + j;
    if (l < L_) {
      const float a = accFC[j] + fc0_b[l];
      wAct[m_my * L_ + l] = a > 0.f ? a : 0.f;
    }
  }
  asm volatile("" ::: "memory");

  // output layer: 16*612 results per wave, all operands in LDS
  for (int idx = lane; idx < TILE * O_; idx += 32) {
    const int m = idx / O_;
    const int o = idx - m * O_;
    float s = sOB[o];
    #pragma unroll
    for (int l = 0; l < L_; ++l) s += wAct[m * L_ + l] * sOW[o * L_ + l];
    out[(size_t)(tileBase + m) * O_ + o] = s;
  }
}

extern "C" void kernel_launch(void* const* d_in, const int* in_sizes, int n_in,
                              void* d_out, int out_size, void* d_ws, size_t ws_size,
                              hipStream_t stream) {
  (void)in_sizes; (void)n_in; (void)d_ws; (void)ws_size; (void)out_size;
  const float* x     = (const float*)d_in[0];
  const float* W_ih  = (const float*)d_in[1];
  const float* W_hh  = (const float*)d_in[2];
  const float* b_ih  = (const float*)d_in[3];
  const float* b_hh  = (const float*)d_in[4];
  const float* fc0_w = (const float*)d_in[5];
  const float* fc0_b = (const float*)d_in[6];
  const float* out_w = (const float*)d_in[7];
  const float* out_b = (const float*)d_in[8];
  float* out = (float*)d_out;

  dim3 grid(B_ / (WAVES * TILE));   // 128 blocks
  dim3 block(128);                  // 4 wave32 per block
  hipLaunchKernelGGL(lstm_fused_kernel, grid, block, 0, stream,
                     x, W_ih, W_hh, b_ih, b_hh, fc0_w, fc0_b, out_w, out_b, out);
}
